// Decoder_Block_40235253629312
// MI455X (gfx1250) — matrix operand order repaired and checked by reading, not compiled, not run on hardware
//
#include <hip/hip_runtime.h>
#include <math.h>

// ---------------------------------------------------------------------------
// Decoder block for MI455X (gfx1250, wave32, WMMA).
// All big-tensor GEMMs run as bf16 WMMA (16x16x32) with f32 accumulation;
// LayerNorm / bias / residual / ReLU / qk-scaling fuse into GEMM epilogues so
// the 256MB edge tensor makes the minimum number of HBM passes (~2.6GB total,
// ~115us at 23.3TB/s -> memory bound; bf16 WMMA compute ~8M instrs, well
// under that roof).
//
// d_ws requirement: ~270 MB (ma2 intermediate 256MB + ~12MB small buffers).
// d_out's ma region (256MB) is reused as scratch for attn/hidden before the
// final in-place write.
// ---------------------------------------------------------------------------

#define C_DIM 256
#define AST   264          // bf16 elements per A row in LDS (padded)
#define OST   260          // f32 elements per row in LDS o-buffer (padded)
#define QK_SCALE (1.0f / 16.0f)   // 1/sqrt(256)

typedef __bf16 bf16_t;
typedef __attribute__((ext_vector_type(16))) __bf16        v16bf;
typedef __attribute__((ext_vector_type(8)))  float         v8f;
typedef __attribute__((ext_vector_type(4)))  unsigned int  vu4;

union Frag16 { v16bf v; struct { vu4 lo; vu4 hi; } u; };

// ---- cooperative loaders ---------------------------------------------------

// Load 256x256 f32 weight (row-major, K x N) into LDS as bf16, pre-swizzled
// into WMMA B-fragment order: for (ktile,ntile) block, lane = (n&15)+16*(kin>>4),
// element e = kin&15; each lane's 16 B values are contiguous (2x b128 loads).
__device__ __forceinline__ void load_w_lds(bf16_t* sw, const float* __restrict__ W, int t) {
  #pragma unroll 4
  for (int i = 0; i < 64; ++i) {
    int vidx = i * 256 + t;                       // 16384 float4 total
    float4 f = ((const float4*)W)[vidx];
    int k  = vidx >> 6;
    int n0 = (vidx & 63) << 2;
    int ktile = k >> 5, kin = k & 31;
    int e = kin & 15;
    int lanebase = (kin >> 4) << 4;
    float vals[4] = {f.x, f.y, f.z, f.w};
    #pragma unroll
    for (int u = 0; u < 4; ++u) {
      int n = n0 + u;
      int lane = lanebase + (n & 15);
      sw[((((ktile * 16 + (n >> 4)) * 32) + lane) << 4) + e] = (bf16_t)vals[u];
    }
  }
}

// Load 16 rows x 256 f32 -> bf16 A tile (row-major, stride AST).
__device__ __forceinline__ void load_a_plain(bf16_t* sa, const float* __restrict__ src, int t) {
  #pragma unroll
  for (int i = 0; i < 4; ++i) {
    int vidx = i * 256 + t;                       // 1024 float4
    float4 f = ((const float4*)src)[vidx];
    int r  = vidx >> 6;
    int c0 = (vidx & 63) << 2;
    bf16_t* p = sa + r * AST + c0;
    p[0] = (bf16_t)f.x; p[1] = (bf16_t)f.y; p[2] = (bf16_t)f.z; p[3] = (bf16_t)f.w;
  }
}

// Load 16 rows x 256 f32 into LDS f32 buffer (stride OST).
__device__ __forceinline__ void load_rows_f32(float* so, const float* __restrict__ src, int t) {
  #pragma unroll
  for (int i = 0; i < 4; ++i) {
    int vidx = i * 256 + t;
    float4 f = ((const float4*)src)[vidx];
    int r  = vidx >> 6;
    int c0 = (vidx & 63) << 2;
    *(float4*)(so + r * OST + c0) = f;
  }
}

// Per-row mean / rstd over 16 rows x 256 cols in so. Caller must have synced.
__device__ __forceinline__ void row_stats(const float* so, float* redS, float* redQ,
                                          float* mus, float* rstds, int t) {
  int r = t >> 4, g = t & 15;
  float s = 0.f, q = 0.f;
  const float* row = so + r * OST + g * 16;
  #pragma unroll
  for (int i = 0; i < 16; ++i) { float x = row[i]; s += x; q += x * x; }
  redS[r * 16 + g] = s;
  redQ[r * 16 + g] = q;
  __syncthreads();
  if (t < 16) {
    float ss = 0.f, qq = 0.f;
    #pragma unroll
    for (int i = 0; i < 16; ++i) { ss += redS[t * 16 + i]; qq += redQ[t * 16 + i]; }
    float mu  = ss * (1.0f / 256.0f);
    float var = qq * (1.0f / 256.0f) - mu * mu;
    mus[t]   = mu;
    rstds[t] = rsqrtf(var + 1e-5f);
  }
  __syncthreads();
}

// Apply row LN from o-buffer and store 16x256 f32 to global.
__device__ __forceinline__ void store_ln_rows(float* __restrict__ dst, const float* so,
                                              const float* mus, const float* rstds,
                                              const float* __restrict__ g,
                                              const float* __restrict__ b, int t) {
  int r = t >> 4, q = t & 15;
  float mu = mus[r], rs = rstds[r];
  const float* row = so + r * OST + q * 16;
  float* d = dst + r * 256 + q * 16;
  #pragma unroll
  for (int i = 0; i < 16; ++i) {
    int c = q * 16 + i;
    d[i] = (row[i] - mu) * rs * g[c] + b[c];
  }
}

// 16-row x 256-col row block GEMM: acc[c] (+)= A(16x256) * W(256 x [32wv..32wv+32))
__device__ __forceinline__ void gemm_rowblock(const bf16_t* sa, const bf16_t* sw,
                                              int lane, int wv, v8f* acc) {
  const int m = lane & 15, kh = lane >> 4;
  #pragma unroll
  for (int kt = 0; kt < 8; ++kt) {
    Frag16 a;
    const bf16_t* ap = sa + m * AST + kt * 32 + kh * 16;
    a.u.lo = *(const vu4*)ap;
    a.u.hi = *(const vu4*)(ap + 8);
    #pragma unroll
    for (int c = 0; c < 2; ++c) {
      Frag16 bm;
      const bf16_t* bp = sw + (((kt * 16 + (wv * 2 + c)) * 32 + lane) << 4);
      bm.u.lo = *(const vu4*)bp;
      bm.u.hi = *(const vu4*)(bp + 8);
      acc[c] = __builtin_amdgcn_wmma_f32_16x16x32_bf16(
          false, a.v, false, bm.v, (short)0, acc[c], false, false);
    }
  }
}

// ---- kernels ---------------------------------------------------------------

// y = LN(x)*g + b, rows = nblocks*16
__global__ __launch_bounds__(256) void k_ln_rows(const float* __restrict__ x,
                                                 const float* __restrict__ g,
                                                 const float* __restrict__ b,
                                                 float* __restrict__ y, int nblocks) {
  __shared__ float so[16 * OST];
  __shared__ float redS[256], redQ[256], mus[16], rstds[16];
  int t = threadIdx.x;
  for (int blk = blockIdx.x; blk < nblocks; blk += gridDim.x) {
    load_rows_f32(so, x + (size_t)blk * 4096, t);
    __syncthreads();
    row_stats(so, redS, redQ, mus, rstds, t);
    store_ln_rows(y + (size_t)blk * 4096, so, mus, rstds, g, b, t);
    __syncthreads();
  }
}

// Y = A @ W + bias (optional ReLU)
__global__ __launch_bounds__(256) void k_gemm_bias(const float* __restrict__ A,
                                                   const float* __restrict__ W,
                                                   const float* __restrict__ bias,
                                                   float* __restrict__ Y,
                                                   int nblocks, int relu) {
  __shared__ bf16_t sw[65536];
  __shared__ bf16_t sa[16 * AST];
  int t = threadIdx.x, lane = t & 31, wv = t >> 5;
  load_w_lds(sw, W, t);
  __syncthreads();
  for (int blk = blockIdx.x; blk < nblocks; blk += gridDim.x) {
    load_a_plain(sa, A + (size_t)blk * 4096, t);
    __syncthreads();
    v8f acc[2] = {{0,0,0,0,0,0,0,0}, {0,0,0,0,0,0,0,0}};
    gemm_rowblock(sa, sw, lane, wv, acc);
    int nin = lane & 15, kh = lane >> 4;
    #pragma unroll
    for (int c = 0; c < 2; ++c) {
      int n = (wv * 2 + c) * 16 + nin;
      float bv = bias[n];
      #pragma unroll
      for (int r = 0; r < 8; ++r) {
        int m = r + 8 * kh;
        float v = acc[c][r] + bv;
        if (relu) v = fmaxf(v, 0.0f);
        Y[(size_t)blk * 4096 + m * 256 + n] = v;
      }
    }
    __syncthreads();
  }
}

// Y = LN(A @ W + bias + R, g2, b2)
__global__ __launch_bounds__(256) void k_gemm_bias_res_ln(const float* __restrict__ A,
                                                          const float* __restrict__ W,
                                                          const float* __restrict__ bias,
                                                          const float* __restrict__ R,
                                                          const float* __restrict__ g2,
                                                          const float* __restrict__ b2,
                                                          float* __restrict__ Y, int nblocks) {
  __shared__ bf16_t sw[65536];
  __shared__ bf16_t sa[16 * AST];
  __shared__ float  so[16 * OST];
  __shared__ float  redS[256], redQ[256], mus[16], rstds[16];
  int t = threadIdx.x, lane = t & 31, wv = t >> 5;
  load_w_lds(sw, W, t);
  __syncthreads();
  for (int blk = blockIdx.x; blk < nblocks; blk += gridDim.x) {
    load_a_plain(sa, A + (size_t)blk * 4096, t);
    load_rows_f32(so, R + (size_t)blk * 4096, t);
    __syncthreads();
    v8f acc[2] = {{0,0,0,0,0,0,0,0}, {0,0,0,0,0,0,0,0}};
    gemm_rowblock(sa, sw, lane, wv, acc);
    int nin = lane & 15, kh = lane >> 4;
    #pragma unroll
    for (int c = 0; c < 2; ++c) {
      int n = (wv * 2 + c) * 16 + nin;
      float bv = bias[n];
      #pragma unroll
      for (int r = 0; r < 8; ++r) {
        int m = r + 8 * kh;
        so[m * OST + n] += acc[c][r] + bv;
      }
    }
    __syncthreads();
    row_stats(so, redS, redQ, mus, rstds, t);
    store_ln_rows(Y + (size_t)blk * 4096, so, mus, rstds, g2, b2, t);
    __syncthreads();
  }
}

// attn = (LN(mol_adj, ln1_ma) @ Wvma + bvma) * q2[b,i,:] * k2[j,:] * scale
__global__ __launch_bounds__(256) void k_big_attn(const float* __restrict__ mol_adj,
                                                  const float* __restrict__ g1,
                                                  const float* __restrict__ b1,
                                                  const float* __restrict__ W,
                                                  const float* __restrict__ bias,
                                                  const float* __restrict__ q2,
                                                  const float* __restrict__ k2,
                                                  float* __restrict__ attn, int nblocks) {
  __shared__ bf16_t sw[65536];
  __shared__ bf16_t sa[16 * AST];
  __shared__ float  so[16 * OST];
  __shared__ float  redS[256], redQ[256], mus[16], rstds[16];
  __shared__ float  qrow[256];
  int t = threadIdx.x, lane = t & 31, wv = t >> 5;
  load_w_lds(sw, W, t);
  __syncthreads();
  for (int blk = blockIdx.x; blk < nblocks; blk += gridDim.x) {
    load_rows_f32(so, mol_adj + (size_t)blk * 4096, t);
    qrow[t] = q2[(size_t)(blk >> 3) * 256 + t];   // 16 rows share (b,i)
    __syncthreads();
    row_stats(so, redS, redQ, mus, rstds, t);
    {   // normalize rows into bf16 A tile
      int r = t >> 4, q = t & 15;
      float mu = mus[r], rs = rstds[r];
      const float* row = so + r * OST + q * 16;
      bf16_t* dst = sa + r * AST + q * 16;
      #pragma unroll
      for (int i = 0; i < 16; ++i) {
        int c = q * 16 + i;
        dst[i] = (bf16_t)((row[i] - mu) * rs * g1[c] + b1[c]);
      }
    }
    __syncthreads();
    v8f acc[2] = {{0,0,0,0,0,0,0,0}, {0,0,0,0,0,0,0,0}};
    gemm_rowblock(sa, sw, lane, wv, acc);
    int nin = lane & 15, kh = lane >> 4;
    int jbase = (blk & 7) * 16;
    #pragma unroll
    for (int c = 0; c < 2; ++c) {
      int n = (wv * 2 + c) * 16 + nin;
      float bv = bias[n], qv = qrow[n];
      #pragma unroll
      for (int r = 0; r < 8; ++r) {
        int m = r + 8 * kh;
        int j = jbase + m;
        float v = (acc[c][r] + bv) * qv * k2[j * 256 + n] * QK_SCALE;
        attn[(size_t)blk * 4096 + m * 256 + n] = v;
      }
    }
    __syncthreads();
  }
}

// ma2 = LN( LN(mol_adj,ln1_ma) + attn @ Wouted + bouted, ln3_ma )
__global__ __launch_bounds__(256) void k_big_edge_out(const float* __restrict__ attn,
                                                      const float* __restrict__ mol_adj,
                                                      const float* __restrict__ g1,
                                                      const float* __restrict__ b1,
                                                      const float* __restrict__ W,
                                                      const float* __restrict__ bias,
                                                      const float* __restrict__ g3,
                                                      const float* __restrict__ b3,
                                                      float* __restrict__ ma2, int nblocks) {
  __shared__ bf16_t sw[65536];
  __shared__ bf16_t sa[16 * AST];
  __shared__ float  so[16 * OST];
  __shared__ float  redS[256], redQ[256], mus[16], rstds[16];
  int t = threadIdx.x, lane = t & 31, wv = t >> 5;
  load_w_lds(sw, W, t);
  __syncthreads();
  for (int blk = blockIdx.x; blk < nblocks; blk += gridDim.x) {
    load_a_plain(sa, attn + (size_t)blk * 4096, t);
    load_rows_f32(so, mol_adj + (size_t)blk * 4096, t);
    __syncthreads();
    row_stats(so, redS, redQ, mus, rstds, t);
    {   // normalize residual rows (ln1_ma) in place
      int r = t >> 4, q = t & 15;
      float mu = mus[r], rs = rstds[r];
      float* row = so + r * OST + q * 16;
      #pragma unroll
      for (int i = 0; i < 16; ++i) {
        int c = q * 16 + i;
        row[i] = (row[i] - mu) * rs * g1[c] + b1[c];
      }
    }
    __syncthreads();
    v8f acc[2] = {{0,0,0,0,0,0,0,0}, {0,0,0,0,0,0,0,0}};
    gemm_rowblock(sa, sw, lane, wv, acc);
    int nin = lane & 15, kh = lane >> 4;
    #pragma unroll
    for (int c = 0; c < 2; ++c) {
      int n = (wv * 2 + c) * 16 + nin;
      float bv = bias[n];
      #pragma unroll
      for (int r = 0; r < 8; ++r) {
        int m = r + 8 * kh;
        so[m * OST + n] += acc[c][r] + bv;
      }
    }
    __syncthreads();
    row_stats(so, redS, redQ, mus, rstds, t);
    store_ln_rows(ma2 + (size_t)blk * 4096, so, mus, rstds, g3, b3, t);
    __syncthreads();
  }
}

// vo[b,i,c] = sum_j softmax_j(attn[b,i,j,c]) * v2[b,j,c]   (online softmax)
__global__ __launch_bounds__(256) void k_softmax_vo(const float* __restrict__ attn,
                                                    const float* __restrict__ v2,
                                                    float* __restrict__ vo) {
  int bi = blockIdx.x;            // 0..2047 == b*128 + i
  int t  = threadIdx.x;           // channel
  const float* ap = attn + (size_t)bi * 128 * 256 + t;
  const float* vp = v2 + (size_t)(bi >> 7) * 128 * 256 + t;
  float mx = -3.0e38f, s = 0.f, ws = 0.f;
  for (int j = 0; j < 128; ++j) {
    float a  = ap[(size_t)j * 256];
    float vv = vp[(size_t)j * 256];
    float m2   = fmaxf(mx, a);
    float corr = __expf(mx - m2);
    float e    = __expf(a - m2);
    s  = s * corr + e;
    ws = ws * corr + e * vv;
    mx = m2;
  }
  vo[(size_t)bi * 256 + t] = ws / s;
}

// protein self-attention: o[i,c] = sum_j softmax_j(q[i,c]*k[j,c]*scale) * v[j,c]
__global__ __launch_bounds__(256) void k_prot_attn(const float* __restrict__ q,
                                                   const float* __restrict__ k,
                                                   const float* __restrict__ v,
                                                   float* __restrict__ o) {
  int i = blockIdx.x, t = threadIdx.x;
  float qv = q[(size_t)i * 256 + t] * QK_SCALE;
  float mx = -3.0e38f, s = 0.f, ws = 0.f;
  for (int j = 0; j < 128; ++j) {
    float a  = qv * k[(size_t)j * 256 + t];
    float vv = v[(size_t)j * 256 + t];
    float m2   = fmaxf(mx, a);
    float corr = __expf(mx - m2);
    float e    = __expf(a - m2);
    s  = s * corr + e;
    ws = ws * corr + e * vv;
    mx = m2;
  }
  o[(size_t)i * 256 + t] = ws / s;
}

// ---------------------------------------------------------------------------

extern "C" void kernel_launch(void* const* d_in, const int* in_sizes, int n_in,
                              void* d_out, int out_size, void* d_ws, size_t ws_size,
                              hipStream_t stream) {
  (void)in_sizes; (void)n_in; (void)out_size; (void)ws_size;

  // inputs flattened in setup_inputs() dict insertion order (nested dicts too)
  const float* mol_annot = (const float*)d_in[0];
  const float* mol_adj   = (const float*)d_in[1];
  const float* prot      = (const float*)d_in[2];
  const float* ln1mx_g = (const float*)d_in[3],  *ln1mx_b = (const float*)d_in[4];
  const float* ln1ma_g = (const float*)d_in[5],  *ln1ma_b = (const float*)d_in[6];
  const float* ln1px_g = (const float*)d_in[7],  *ln1px_b = (const float*)d_in[8];
  const float* ln2px_g = (const float*)d_in[9],  *ln2px_b = (const float*)d_in[10];
  const float* ln3ma_g = (const float*)d_in[11], *ln3ma_b = (const float*)d_in[12];
  const float* ln3mx_g = (const float*)d_in[13], *ln3mx_b = (const float*)d_in[14];
  const float* ln4ma_g = (const float*)d_in[15], *ln4ma_b = (const float*)d_in[16];
  const float* ln4mx_g = (const float*)d_in[17], *ln4mx_b = (const float*)d_in[18];
  const float* pq_w  = (const float*)d_in[19], *pq_b  = (const float*)d_in[20];
  const float* pk_w  = (const float*)d_in[21], *pk_b  = (const float*)d_in[22];
  const float* pv_w  = (const float*)d_in[23], *pv_b  = (const float*)d_in[24];
  const float* po_w  = (const float*)d_in[25], *po_b  = (const float*)d_in[26];
  const float* qmx_w = (const float*)d_in[27], *qmx_b = (const float*)d_in[28];
  const float* kpx_w = (const float*)d_in[29], *kpx_b = (const float*)d_in[30];
  const float* vmx_w = (const float*)d_in[31], *vmx_b = (const float*)d_in[32];
  const float* vma_w = (const float*)d_in[33], *vma_b = (const float*)d_in[34];
  const float* ond_w = (const float*)d_in[35], *ond_b = (const float*)d_in[36];
  const float* oed_w = (const float*)d_in[37], *oed_b = (const float*)d_in[38];
  const float* mafc1_w = (const float*)d_in[39], *mafc1_b = (const float*)d_in[40];
  const float* mafc2_w = (const float*)d_in[41], *mafc2_b = (const float*)d_in[42];
  const float* mxfc1_w = (const float*)d_in[43], *mxfc1_b = (const float*)d_in[44];
  const float* mxfc2_w = (const float*)d_in[45], *mxfc2_b = (const float*)d_in[46];

  float* out_mx = (float*)d_out;               // 16*128*256
  float* out_ma = out_mx + 524288;             // 16*128*128*256 (also scratch)

  // workspace layout (floats)
  float* wsf = (float*)d_ws;
  float* MA2 = wsf;                            // 67108864
  float* PX  = wsf + 67108864;                 // 32768
  float* QP  = PX  + 32768;
  float* KP  = QP  + 32768;
  float* VP  = KP  + 32768;
  float* OP  = VP  + 32768;
  float* PX2 = OP  + 32768;
  float* K2B = PX2 + 32768;
  float* MX  = K2B + 32768;                    // 524288
  float* Q2B = MX  + 524288;
  float* V2B = Q2B + 524288;
  float* VO  = V2B + 524288;
  float* MX2 = VO  + 524288;
  float* HMX = MX2 + 524288;

  dim3 blk(256);

  // --- protein path (tiny) ---
  k_ln_rows<<<8, blk, 0, stream>>>(prot, ln1px_g, ln1px_b, PX, 8);
  k_gemm_bias<<<8, blk, 0, stream>>>(PX, pq_w, pq_b, QP, 8, 0);
  k_gemm_bias<<<8, blk, 0, stream>>>(PX, pk_w, pk_b, KP, 8, 0);
  k_gemm_bias<<<8, blk, 0, stream>>>(PX, pv_w, pv_b, VP, 8, 0);
  k_prot_attn<<<128, blk, 0, stream>>>(QP, KP, VP, OP);
  k_gemm_bias_res_ln<<<8, blk, 0, stream>>>(OP, po_w, po_b, PX, ln2px_g, ln2px_b, PX2, 8);
  k_gemm_bias<<<8, blk, 0, stream>>>(PX2, kpx_w, kpx_b, K2B, 8, 0);

  // --- node path projections ---
  k_ln_rows<<<128, blk, 0, stream>>>(mol_annot, ln1mx_g, ln1mx_b, MX, 128);
  k_gemm_bias<<<128, blk, 0, stream>>>(MX, qmx_w, qmx_b, Q2B, 128, 0);
  k_gemm_bias<<<128, blk, 0, stream>>>(MX, vmx_w, vmx_b, V2B, 128, 0);

  // --- edge tensor (memory-bound big passes) ---
  k_big_attn<<<2048, blk, 0, stream>>>(mol_adj, ln1ma_g, ln1ma_b, vma_w, vma_b,
                                       Q2B, K2B, out_ma, 16384);
  k_big_edge_out<<<2048, blk, 0, stream>>>(out_ma, mol_adj, ln1ma_g, ln1ma_b,
                                           oed_w, oed_b, ln3ma_g, ln3ma_b, MA2, 16384);
  k_softmax_vo<<<2048, blk, 0, stream>>>(out_ma, V2B, VO);

  // --- node output path ---
  k_gemm_bias_res_ln<<<128, blk, 0, stream>>>(VO, ond_w, ond_b, MX, ln3mx_g, ln3mx_b, MX2, 128);
  k_gemm_bias<<<128, blk, 0, stream>>>(MX2, mxfc1_w, mxfc1_b, HMX, 128, 1);
  k_gemm_bias_res_ln<<<128, blk, 0, stream>>>(HMX, mxfc2_w, mxfc2_b, MX2,
                                              ln4mx_g, ln4mx_b, out_mx, 128);

  // --- edge MLP (hidden reuses out_ma region; final write in place) ---
  k_gemm_bias<<<2048, blk, 0, stream>>>(MA2, mafc1_w, mafc1_b, out_ma, 16384, 1);
  k_gemm_bias_res_ln<<<2048, blk, 0, stream>>>(out_ma, mafc2_w, mafc2_b, MA2,
                                               ln4ma_g, ln4ma_b, out_ma, 16384);
}
